// GIN_51677046505640
// MI455X (gfx1250) — compile-verified, matrix operand order checked by reading, not run on hardware
//
#include <hip/hip_runtime.h>

#define N_NODES  100000
#define N_EDGES  1600000
#define N_FEAT   128
#define N_GRAPHS 100

typedef __attribute__((ext_vector_type(16))) _Float16 v16h;
typedef __attribute__((ext_vector_type(8)))  _Float16 v8h;
typedef __attribute__((ext_vector_type(8)))  float    v8f;
typedef __attribute__((ext_vector_type(4)))  int      v4i;

typedef __attribute__((address_space(1))) v4i gv4i;   // global int4
typedef __attribute__((address_space(3))) v4i lv4i;   // LDS int4

#if defined(__gfx1250__) && __has_builtin(__builtin_amdgcn_global_load_async_to_lds_b128)
#define HAVE_ASYNC_LDS 1
#else
#define HAVE_ASYNC_LDS 0
#endif

constexpr int THREADS        = 128;            // 4 wave32 per block
constexpr int WAVES          = THREADS / 32;
constexpr int ROWS_PER_BLOCK = WAVES * 16;     // 64 rows of M per block
constexpr int LSTR           = 136;            // LDS row stride (halves): conflict-free b128 frag loads
constexpr int WMAT           = 128 * 128;      // elements per weight matrix

__device__ inline void wait_async0() {
#if defined(__gfx1250__)
#if __has_builtin(__builtin_amdgcn_s_wait_asynccnt)
    __builtin_amdgcn_s_wait_asynccnt(0);
#else
    asm volatile("s_wait_asynccnt 0" ::: "memory");
#endif
#endif
}

// Stage a pre-converted f16 weight matrix [128][128] into padded LDS [128][LSTR].
// Async path: per-lane 16B global->LDS DMA (ASYNCcnt); fallback: v8h copies.
__device__ inline void stage_weights(const _Float16* __restrict__ Wg,
                                     _Float16* sW, int tid) {
    for (int i = tid; i < 128 * 16; i += THREADS) {
        const int r = i >> 4, c8 = (i & 15) << 3;   // 8 halves = 16B per transfer
#if HAVE_ASYNC_LDS
        const v4i* gp = (const v4i*)(Wg + r * 128 + c8);
        v4i*       lp = (v4i*)(&sW[r * LSTR + c8]);
        __builtin_amdgcn_global_load_async_to_lds_b128((gv4i*)gp, (lv4i*)lp, 0, 0);
#else
        *(v8h*)(&sW[r * LSTR + c8]) = *(const v8h*)(Wg + r * 128 + c8);
#endif
    }
}

// Load a 16x32 f16 WMMA operand fragment (ISA 7.12.2 16-bit A layout; B uses the
// transposed-identical mapping since W is held row-major as [n][k]).
__device__ inline v16h load_frag(const _Float16* __restrict__ rowbase, int lane, int ktile) {
    const int r  = lane & 15;
    const int kb = (lane & 16) ? 8 : 0;
    const _Float16* p = rowbase + r * LSTR + ktile * 32 + kb;
    v8h lo = *(const v8h*)(p);
    v8h hi = *(const v8h*)(p + 16);
    v16h out;
#pragma unroll
    for (int i = 0; i < 8; ++i) { out[i] = lo[i]; out[i + 8] = hi[i]; }
    return out;
}

// Convert the 4 f32 weight matrices to f16 once (64K elements total).
__global__ __launch_bounds__(256) void gin_wcvt(
    const float* __restrict__ W1, const float* __restrict__ W2,
    const float* __restrict__ W3, const float* __restrict__ W4,
    _Float16* __restrict__ out)
{
    const int i = blockIdx.x * 256 + threadIdx.x;
    if (i >= 4 * WMAT) return;
    const float* Ws[4] = {W1, W2, W3, W4};
    out[i] = (_Float16)Ws[i >> 14][i & (WMAT - 1)];
}

// Fused 2-layer MLP: out = relu(relu(in*Wa^T + ba)*Wb^T + bb).
// Each wave owns a 16-row M tile; layer-1 result stays in LDS (f16).
__global__ __launch_bounds__(THREADS) void gin_mlp2(
    const float* __restrict__ in,
    const _Float16* __restrict__ Wa16, const float* __restrict__ ba,
    const _Float16* __restrict__ Wb16, const float* __restrict__ bb,
    float* __restrict__ out)
{
    __shared__ _Float16 sW[128 * LSTR];              // current weight matrix, f16 [n][k]
    __shared__ _Float16 sA[ROWS_PER_BLOCK * LSTR];   // activations, f16

    const int tid      = threadIdx.x;
    const int lane     = tid & 31;
    const int wave     = tid >> 5;
    const int blockRow = blockIdx.x * ROWS_PER_BLOCK;

    // Warm L2 with the second weight matrix (global_prefetch_b8): 128 thr * 256B = 32KB.
    __builtin_prefetch(Wb16 + tid * 128, 0, 1);

    // Stage W1 into LDS (async DMA on gfx1250).
    stage_weights(Wa16, sW, tid);

    // Stage this block's activation rows (f32 -> f16) into LDS.
    for (int i = tid; i < ROWS_PER_BLOCK * 32; i += THREADS) {
        const int r = i >> 5, c4 = (i & 31) << 2;
        const int row = blockRow + r;
        float4 v = make_float4(0.f, 0.f, 0.f, 0.f);
        if (row < N_NODES) v = *(const float4*)(in + (size_t)row * N_FEAT + c4);
        _Float16* q = &sA[r * LSTR + c4];
        q[0] = (_Float16)v.x; q[1] = (_Float16)v.y; q[2] = (_Float16)v.z; q[3] = (_Float16)v.w;
    }
#if HAVE_ASYNC_LDS
    wait_async0();
#endif
    __syncthreads();

    const _Float16* aBase = &sA[wave * 16 * LSTR];

    // ---------------- layer 1 ----------------
    v8f acc[8];
#pragma unroll
    for (int nt = 0; nt < 8; ++nt) {
        const float bv = ba[nt * 16 + (lane & 15)];
        v8f c;
#pragma unroll
        for (int i = 0; i < 8; ++i) c[i] = bv;
        acc[nt] = c;
    }
#pragma unroll
    for (int kt = 0; kt < 4; ++kt) {
        const v16h a = load_frag(aBase, lane, kt);
#pragma unroll
        for (int nt = 0; nt < 8; ++nt) {
            const v16h b = load_frag(&sW[nt * 16 * LSTR], lane, kt);
            acc[nt] = __builtin_amdgcn_wmma_f32_16x16x32_f16(
                false, a, false, b, (short)0, acc[nt], false, false);
        }
    }
    // ReLU; write h1 back (f16) into this wave's own sA rows (no cross-wave sharing).
#pragma unroll
    for (int nt = 0; nt < 8; ++nt) {
#pragma unroll
        for (int r = 0; r < 8; ++r) {
            float v = acc[nt][r];
            v = v > 0.f ? v : 0.f;
            const int mloc = wave * 16 + ((lane & 16) ? 8 : 0) + r;
            sA[mloc * LSTR + nt * 16 + (lane & 15)] = (_Float16)v;
        }
    }
    __syncthreads();   // everyone done reading W1

    // Stage W2 into LDS (reuse sW).
    stage_weights(Wb16, sW, tid);
#if HAVE_ASYNC_LDS
    wait_async0();
#endif
    __syncthreads();

    // ---------------- layer 2 ----------------
#pragma unroll
    for (int nt = 0; nt < 8; ++nt) {
        const float bv = bb[nt * 16 + (lane & 15)];
        v8f c;
#pragma unroll
        for (int i = 0; i < 8; ++i) c[i] = bv;
        acc[nt] = c;
    }
#pragma unroll
    for (int kt = 0; kt < 4; ++kt) {
        const v16h a = load_frag(aBase, lane, kt);
#pragma unroll
        for (int nt = 0; nt < 8; ++nt) {
            const v16h b = load_frag(&sW[nt * 16 * LSTR], lane, kt);
            acc[nt] = __builtin_amdgcn_wmma_f32_16x16x32_f16(
                false, a, false, b, (short)0, acc[nt], false, false);
        }
    }
    // ReLU + store f32 to global.
#pragma unroll
    for (int nt = 0; nt < 8; ++nt) {
#pragma unroll
        for (int r = 0; r < 8; ++r) {
            float v = acc[nt][r];
            v = v > 0.f ? v : 0.f;
            const int mrow = blockRow + wave * 16 + ((lane & 16) ? 8 : 0) + r;
            if (mrow < N_NODES)
                out[(size_t)mrow * N_FEAT + nt * 16 + (lane & 15)] = v;
        }
    }
}

// agg[dst] += feat[src], warp per edge, float4 per lane (128 feats / 32 lanes).
__global__ __launch_bounds__(256) void gin_scatter_add(
    const float* __restrict__ feat,
    const int* __restrict__ esrc, const int* __restrict__ edst,
    float* __restrict__ agg)
{
    const int e    = (blockIdx.x * blockDim.x + threadIdx.x) >> 5;
    const int lane = threadIdx.x & 31;
    if (e >= N_EDGES) return;
    const int s = esrc[e], d = edst[e];
    const float4 v = *(const float4*)(feat + (size_t)s * N_FEAT + lane * 4);
    float* o = agg + (size_t)d * N_FEAT + lane * 4;
    atomicAdd(o + 0, v.x); atomicAdd(o + 1, v.y);
    atomicAdd(o + 2, v.z); atomicAdd(o + 3, v.w);
}

__global__ __launch_bounds__(256) void gin_copy(const float* __restrict__ in,
                                                float* __restrict__ out, int n4)
{
    const int i = blockIdx.x * blockDim.x + threadIdx.x;
    if (i < n4) ((float4*)out)[i] = ((const float4*)in)[i];
}

__global__ __launch_bounds__(256) void gin_pool_zero(float* __restrict__ sums,
                                                     float* __restrict__ counts)
{
    const int i = blockIdx.x * blockDim.x + threadIdx.x;
    if (i < N_GRAPHS * N_FEAT) sums[i] = 0.f;
    if (i < N_GRAPHS) counts[i] = 0.f;
}

__global__ __launch_bounds__(256) void gin_pool_acc(
    const float* __restrict__ h, const int* __restrict__ batch,
    float* __restrict__ sums, float* __restrict__ counts)
{
    const int n    = (blockIdx.x * blockDim.x + threadIdx.x) >> 5;
    const int lane = threadIdx.x & 31;
    if (n >= N_NODES) return;
    const int g = batch[n];
    const float4 v = *(const float4*)(h + (size_t)n * N_FEAT + lane * 4);
    float* o = sums + (size_t)g * N_FEAT + lane * 4;
    atomicAdd(o + 0, v.x); atomicAdd(o + 1, v.y);
    atomicAdd(o + 2, v.z); atomicAdd(o + 3, v.w);
    if (lane == 0) atomicAdd(&counts[g], 1.f);
}

__global__ __launch_bounds__(256) void gin_pool_div(float* __restrict__ sums,
                                                    const float* __restrict__ counts)
{
    const int i = blockIdx.x * blockDim.x + threadIdx.x;
    if (i < N_GRAPHS * N_FEAT) sums[i] /= fmaxf(counts[i >> 7], 1.f);
}

extern "C" void kernel_launch(void* const* d_in, const int* in_sizes, int n_in,
                              void* d_out, int out_size, void* d_ws, size_t ws_size,
                              hipStream_t stream)
{
    const float* x    = (const float*)d_in[0];
    const int*  eidx  = (const int*)d_in[1];      // [2, N_EDGES]
    const int*  batch = (const int*)d_in[2];      // [N_NODES]
    const float* W1 = (const float*)d_in[3]; const float* b1 = (const float*)d_in[4];
    const float* W2 = (const float*)d_in[5]; const float* b2 = (const float*)d_in[6];
    const float* W3 = (const float*)d_in[7]; const float* b3 = (const float*)d_in[8];
    const float* W4 = (const float*)d_in[9]; const float* b4 = (const float*)d_in[10];

    const int* esrc = eidx;
    const int* edst = eidx + N_EDGES;

    float* pool   = (float*)d_out;                          // [100, 128]
    float* h      = (float*)d_out + N_GRAPHS * N_FEAT;      // [100000, 128]
    float* agg    = (float*)d_ws;                           // [100000, 128]
    float* counts = agg + (size_t)N_NODES * N_FEAT;         // [100] (padded to 128)
    _Float16* w16 = (_Float16*)(counts + 128);              // 4 x [128,128] f16
    _Float16* W1h = w16 + 0 * WMAT;
    _Float16* W2h = w16 + 1 * WMAT;
    _Float16* W3h = w16 + 2 * WMAT;
    _Float16* W4h = w16 + 3 * WMAT;

    const int n4       = N_NODES * N_FEAT / 4;
    const int copyGrid = (n4 + 255) / 256;
    const int scatGrid = (N_EDGES * 32 + 255) / 256;
    const int mlpGrid  = (N_NODES + ROWS_PER_BLOCK - 1) / ROWS_PER_BLOCK;

    // Pre-convert weights to f16 (enables async byte-copy weight staging).
    gin_wcvt<<<(4 * WMAT + 255) / 256, 256, 0, stream>>>(W1, W2, W3, W4, w16);

    // ---- GIN layer 1 ----
    gin_copy<<<copyGrid, 256, 0, stream>>>(x, agg, n4);                       // agg = x
    gin_scatter_add<<<scatGrid, 256, 0, stream>>>(x, esrc, edst, agg);        // agg += sum_j x[src]
    gin_mlp2<<<mlpGrid, THREADS, 0, stream>>>(agg, W1h, b1, W2h, b2, h);      // h = MLP1(agg)

    // ---- GIN layer 2 ----
    gin_copy<<<copyGrid, 256, 0, stream>>>(h, agg, n4);                       // agg = h
    gin_scatter_add<<<scatGrid, 256, 0, stream>>>(h, esrc, edst, agg);        // agg += sum_j h[src]
    gin_mlp2<<<mlpGrid, THREADS, 0, stream>>>(agg, W3h, b3, W4h, b4, h);      // h = MLP2(agg)

    // ---- global mean pool ----
    gin_pool_zero<<<(N_GRAPHS * N_FEAT + 255) / 256, 256, 0, stream>>>(pool, counts);
    gin_pool_acc<<<(N_NODES * 32 + 255) / 256, 256, 0, stream>>>(h, batch, pool, counts);
    gin_pool_div<<<(N_GRAPHS * N_FEAT + 255) / 256, 256, 0, stream>>>(pool, counts);
}
